// Projection_40578851013193
// MI455X (gfx1250) — matrix operand order repaired and checked by reading, not compiled, not run on hardware
//
#include <hip/hip_runtime.h>

// ---------------------------------------------------------------------------
// Projection pipeline for MI455X (gfx1250, wave32):
//   rotate -> trilinear scatter (L2 atomics) -> 3x banded-GEMM Gaussian conv
//   via v_wmma_f32_16x16x32_f16 -> fused DRC product projection.
// Workspace: one B*64^3 f32 volume (33.6 MB) + 8 KB precomputed f16 band
// matrix laid out so WMMA A/B fragments are aligned contiguous b128 loads.
// ---------------------------------------------------------------------------

typedef _Float16 v16h __attribute__((ext_vector_type(16)));
typedef _Float16 v8h  __attribute__((ext_vector_type(8)));
typedef float    v8f  __attribute__((ext_vector_type(8)));

#define VOX3  (64 * 64 * 64)

// Normalized Gaussian taps: g[d] = exp(-d^2/18)/Z, Z = 7.5165544 (21 taps, sigma=3)
__device__ __forceinline__ float gw(int d) {
  if (d < -10 || d > 10) return 0.0f;
  float fd = (float)d;
  return __expf(fd * fd * (-1.0f / 18.0f)) * 0.13303965f;
}

__device__ __forceinline__ float clamp01(float v) {
  return fminf(fmaxf(v, 0.0f), 1.0f);
}

// Build v16h from two aligned 16B chunks.
__device__ __forceinline__ v16h join16h(const _Float16* p0, const _Float16* p1) {
  union { v16h v; v8h h[2]; } u;
  u.h[0] = *(const v8h*)p0;
  u.h[1] = *(const v8h*)p1;
  return u.v;
}

// ---------------------------------------------------------------------------
// Kernel 0: materialize symmetric band matrix band[m][k] = g(|m-k|) as f16.
// Runs once per launch; single tiny grid. All expf lives here.
// ---------------------------------------------------------------------------
__global__ void band_init_kernel(_Float16* __restrict__ band) {
  int t = blockIdx.x * blockDim.x + threadIdx.x;   // 0..4095
  int m = t >> 6, k = t & 63;
  band[t] = (_Float16)gw(m - k);
}

// ---------------------------------------------------------------------------
// Kernel 1: rotate points + trilinear scatter into vol[b][z][y][x]
// ---------------------------------------------------------------------------
__global__ void scatter_kernel(const float* __restrict__ pc,
                               const float* __restrict__ rot,
                               float* __restrict__ vol, int npts) {
  int b = blockIdx.y;
  int n = blockIdx.x * blockDim.x + threadIdx.x;
  if (n >= npts) return;
  const float* pp = pc + ((long)b * npts + n) * 3;
  float px = pp[0], py = pp[1], pz = pp[2];
  const float* R = rot + b * 9;
  // pc_t[j] = sum_i pc[i] * rot[j][i]
  float tx = px * R[0] + py * R[1] + pz * R[2];
  float ty = px * R[3] + py * R[4] + pz * R[5];
  float tz = px * R[6] + py * R[7] + pz * R[8];
  float gx = (tx + 0.5f) * 63.0f;
  float gy = (ty + 0.5f) * 63.0f;
  float gz = (tz + 0.5f) * 63.0f;
  float fx = floorf(gx), fy = floorf(gy), fz = floorf(gz);
  int ix = (int)fx, iy = (int)fy, iz = (int)fz;
  float rx = gx - fx, ry = gy - fy, rz = gz - fz;
  float* vb = vol + (long)b * VOX3;
#pragma unroll
  for (int c = 0; c < 8; ++c) {
    int ox = (c >> 2) & 1, oy = (c >> 1) & 1, oz = c & 1;
    int x = ix + ox, y = iy + oy, z = iz + oz;
    if ((unsigned)x < 64u && (unsigned)y < 64u && (unsigned)z < 64u) {
      float w = (ox ? rx : 1.0f - rx) * (oy ? ry : 1.0f - ry) * (oz ? rz : 1.0f - rz);
      atomicAdd(vb + z * 4096 + y * 64 + x, w);
    }
  }
}

// Band matrix as WMMA A fragment: A[m][k], lane row m, K(j)=(j&7)+8*half+16*(j>>3)
// -> two contiguous 8-half runs: [k0+8h .. +7] and [k0+16+8h .. +7].
__device__ __forceinline__ v16h band_a_frag(const _Float16* __restrict__ band,
                                            int m, int k0, int half_) {
  const _Float16* row = band + m * 64;
  return join16h(row + k0 + 8 * half_, row + k0 + 16 + 8 * half_);
}

// Band matrix as WMMA B fragment: B[k][n], lane col n, K(j)=k0+16*half+j.
// Symmetry: B[k][n] = band[n][k] -> one contiguous 16-half run in row n.
__device__ __forceinline__ v16h band_b_frag(const _Float16* __restrict__ band,
                                            int n, int k0, int half_) {
  const _Float16* row = band + n * 64 + k0 + 16 * half_;
  return join16h(row, row + 8);
}

// ---------------------------------------------------------------------------
// Kernel 2: conv along x (contiguous).  C[f,x'] = sum_x clip(V[f,x]) * g[|x-x'|]
// Data rows are WMMA operand A (contiguous K pairs -> float2 loads);
// band matrix is operand B (b128 loads). In place (per-wave disjoint rows).
// ---------------------------------------------------------------------------
__global__ void conv_x_kernel(float* __restrict__ vol,
                              const _Float16* __restrict__ band) {
  int t = threadIdx.x;
  int lane = t & 31, wave = t >> 5;
  int half_ = lane >> 4, lan = lane & 15;
  long fbase = (long)blockIdx.x * 128 + wave * 16;
  const float* rowp = vol + (fbase + lan) * 64;

  v8f acc[4];
#pragma unroll
  for (int nt = 0; nt < 4; ++nt)
#pragma unroll
    for (int e = 0; e < 8; ++e) acc[nt][e] = 0.0f;

#pragma unroll
  for (int k0 = 0; k0 < 64; k0 += 32) {
    // A tile 16x32 f16: lane row = lan; K(j) = (j&7) + 8*half + 16*(j>>3)
    v16h a;
#pragma unroll
    for (int v = 0; v < 8; ++v) {
      int k = k0 + 8 * half_ + ((v >> 2) << 4) + ((v & 3) << 1);
      float2 d = *(const float2*)(rowp + k);
      a[2 * v]     = (_Float16)clamp01(d.x);   // fuse vox clip here (first pass)
      a[2 * v + 1] = (_Float16)clamp01(d.y);
    }
#pragma unroll
    for (int nt = 0; nt < 4; ++nt) {
      v16h bt = band_a_frag(band, nt * 16 + lan, k0, half_);
      acc[nt] = __builtin_amdgcn_wmma_f32_16x16x32_f16(
          false, a, false, bt, (short)0, acc[nt], false, false);
    }
  }
  // C/D layout: lane col n = lan, rows = r + 8*half
  float* outbase = vol + fbase * 64;
#pragma unroll
  for (int nt = 0; nt < 4; ++nt)
#pragma unroll
    for (int r = 0; r < 8; ++r)
      outbase[(r + 8 * half_) * 64 + nt * 16 + lan] = acc[nt][r];
}

// ---------------------------------------------------------------------------
// Kernel 3: conv along y.  Per (b,z) slice S[y][x]: OUT = Band * S.
// Slice staged in LDS (f16, 144B-padded rows -> conflict-free ds_load_b128);
// band = operand A via b128 loads. In place via LDS staging.
// ---------------------------------------------------------------------------
__global__ void conv_y_kernel(float* __restrict__ vol,
                              const _Float16* __restrict__ band) {
  __shared__ __align__(16) _Float16 sb[64][72];  // [x][y]
  int t = threadIdx.x;
  float* slice = vol + (long)blockIdx.x * 4096;
#pragma unroll
  for (int i = 0; i < 32; ++i) {
    int idx = i * 128 + t;
    int n = idx & 63, k = idx >> 6;               // n = x (coalesced), k = y
    sb[n][k] = (_Float16)slice[k * 64 + n];
  }
  __syncthreads();

  int lane = t & 31, wave = t >> 5;
  int half_ = lane >> 4, lan = lane & 15;
  int m0 = wave * 16, m = m0 + lan;               // output row y'

  v8f acc[4];
#pragma unroll
  for (int nt = 0; nt < 4; ++nt)
#pragma unroll
    for (int e = 0; e < 8; ++e) acc[nt][e] = 0.0f;

#pragma unroll
  for (int k0 = 0; k0 < 64; k0 += 32) {
    v16h a = band_a_frag(band, m, k0, half_);
#pragma unroll
    for (int nt = 0; nt < 4; ++nt) {
      const _Float16* p = &sb[nt * 16 + lan][k0 + 8 * half_];
      v16h bt = join16h(p, p + 16);
      acc[nt] = __builtin_amdgcn_wmma_f32_16x16x32_f16(
          false, a, false, bt, (short)0, acc[nt], false, false);
    }
  }
#pragma unroll
  for (int nt = 0; nt < 4; ++nt)
#pragma unroll
    for (int r = 0; r < 8; ++r)
      slice[(m0 + r + 8 * half_) * 64 + nt * 16 + lan] = acc[nt][r];
}

// ---------------------------------------------------------------------------
// Kernel 4: conv along z FUSED with scale, clip and DRC projection:
//   proj[b, 63-y, x] = 1 - prod_z' (1 - clip(conv_z[z',x] * scale[b]))
// Wave-half merge via __shfl_xor(16) (wave32), cross-wave merge via LDS.
// ---------------------------------------------------------------------------
__global__ void conv_z_prod_kernel(const float* __restrict__ vol,
                                   const _Float16* __restrict__ band,
                                   const float* __restrict__ scale,
                                   float* __restrict__ out) {
  __shared__ __align__(16) _Float16 sb[64][72];   // [x][z]
  __shared__ float pp[4][64];
  int t = threadIdx.x;
  int b = blockIdx.x >> 6, y = blockIdx.x & 63;
  const float* basep = vol + (long)b * VOX3 + y * 64;
#pragma unroll
  for (int i = 0; i < 32; ++i) {
    int idx = i * 128 + t;
    int n = idx & 63, k = idx >> 6;               // n = x (coalesced), k = z
    sb[n][k] = (_Float16)basep[(long)k * 4096 + n];
  }
  __syncthreads();

  int lane = t & 31, wave = t >> 5;
  int half_ = lane >> 4, lan = lane & 15;
  int m0 = wave * 16, m = m0 + lan;               // output row z'

  v8f acc[4];
#pragma unroll
  for (int nt = 0; nt < 4; ++nt)
#pragma unroll
    for (int e = 0; e < 8; ++e) acc[nt][e] = 0.0f;

#pragma unroll
  for (int k0 = 0; k0 < 64; k0 += 32) {
    v16h a = band_a_frag(band, m, k0, half_);
#pragma unroll
    for (int nt = 0; nt < 4; ++nt) {
      const _Float16* p = &sb[nt * 16 + lan][k0 + 8 * half_];
      v16h bt = join16h(p, p + 16);
      acc[nt] = __builtin_amdgcn_wmma_f32_16x16x32_f16(
          false, a, false, bt, (short)0, acc[nt], false, false);
    }
  }

  float sc = scale[b];
#pragma unroll
  for (int nt = 0; nt < 4; ++nt) {
    float p = 1.0f;
#pragma unroll
    for (int r = 0; r < 8; ++r) {
      float v = clamp01(acc[nt][r] * sc);
      p *= (1.0f - v);
    }
    p *= __shfl_xor(p, 16, 32);                   // merge z' halves (wave32)
    if (half_ == 0) pp[wave][nt * 16 + lan] = p;  // 16 rows per wave merged
  }
  __syncthreads();
  if (t < 64) {
    float pr = pp[0][t] * pp[1][t] * pp[2][t] * pp[3][t];  // all 64 z'
    out[b * 4096 + (63 - y) * 64 + t] = 1.0f - pr;         // y reversed
  }
}

// ---------------------------------------------------------------------------
extern "C" void kernel_launch(void* const* d_in, const int* in_sizes, int n_in,
                              void* d_out, int out_size, void* d_ws, size_t ws_size,
                              hipStream_t stream) {
  (void)n_in; (void)out_size; (void)ws_size;
  const float* pc    = (const float*)d_in[0];
  const float* rot   = (const float*)d_in[1];
  const float* scale = (const float*)d_in[2];
  int B = in_sizes[2];                 // scale is (B,)
  int N = in_sizes[0] / (3 * B);       // pc is (B,N,3)

  float*    vol  = (float*)d_ws;                       // B*64^3 f32 volume
  _Float16* band = (_Float16*)(vol + (size_t)B * VOX3); // 64*64 f16 band matrix

  hipMemsetAsync(vol, 0, (size_t)B * VOX3 * sizeof(float), stream);
  band_init_kernel<<<32, 128, 0, stream>>>(band);

  dim3 sgrid((N + 255) / 256, B);
  scatter_kernel<<<sgrid, 256, 0, stream>>>(pc, rot, vol, N);

  int fibers = B * 64 * 64;                                     // one per (b,z,y)
  conv_x_kernel<<<fibers / 128, 256, 0, stream>>>(vol, band);   // 8 waves x 16 fibers
  conv_y_kernel<<<B * 64, 128, 0, stream>>>(vol, band);         // one (b,z) slice/block
  conv_z_prod_kernel<<<B * 64, 128, 0, stream>>>(vol, band, scale, (float*)d_out);
}
